// _ScaledDotProductAttention_66700842107097
// MI455X (gfx1250) — matrix operand order repaired and checked by reading, not compiled, not run on hardware
//
#include <hip/hip_runtime.h>

// ---------------------------------------------------------------------------
// Scaled dot-product attention for MI455X (gfx1250, wave32, WMMA).
// B=16, N=2048, D=128, fp32 in, outputs (out[B,N,D], attn[B,N,N]) fp32.
//
// Roofline: attn store (256 MB) dominates -> ~14 us floor @ 23.3 TB/s.
// Design: two-pass online softmax so NO S tile ever lives in registers/scratch
// (the extra QK^T recompute pass is ~17 GFLOP of f16 WMMA, far under the
// store floor). One workgroup = (batch, 16 query rows); 512 threads = 16
// wave32s, wave w owns keys [w*128, (w+1)*128).
//   Pass 1: S tiles via v_wmma_f32_16x16x32_f16 -> online (m, l) per row,
//           merged across lanes (shfl_xor) and waves (LDS).
//   Pass 2: recompute S, p = exp(s-gm)*inv, stream attn to HBM, restage f16 P
//           through LDS (A-layout), GEMM2 P*V with WMMA; partial O reduced
//           across waves with ds_add_f32 atomics.
// ---------------------------------------------------------------------------

typedef __attribute__((ext_vector_type(16))) _Float16 v16h;
typedef __attribute__((ext_vector_type(8)))  _Float16 v8h;
typedef __attribute__((ext_vector_type(8)))  float    v8f;
typedef __attribute__((ext_vector_type(4)))  float    v4f;

#define Bn   16
#define Nn   2048
#define Dn   128
#define NW   16            // waves per block
#define WK   128           // keys per wave
#define NT   (WK / 16)     // 16-key tiles per wave (8)
#define NTHREADS (NW * 32)

// 1/sqrt(128)
#define ATTN_SCALE 0.08838834764831845f

// ---- A-matrix fragment (16x32 f16) from row-major f32:
// lane L: row = L%16, half = L/16; elements hold K = half*16 + 0..15.
// One contiguous 16-float run.
__device__ __forceinline__ v16h frag_a_f32(const float* __restrict__ base,
                                           int ld, int c, int lane) {
  const int row = lane & 15, hf = lane >> 4;
  const float* p = base + (size_t)row * ld + c * 32 + hf * 16;
  v4f a0 = *(const v4f*)(p);
  v4f a1 = *(const v4f*)(p + 4);
  v4f b0 = *(const v4f*)(p + 8);
  v4f b1 = *(const v4f*)(p + 12);
  v16h r;
#pragma unroll
  for (int i = 0; i < 4; ++i) {
    r[i]      = (_Float16)a0[i];
    r[4 + i]  = (_Float16)a1[i];
    r[8 + i]  = (_Float16)b0[i];
    r[12 + i] = (_Float16)b1[i];
  }
  return r;
}

// ---- B-matrix fragment (32x16 f16) for B = K^T, same contiguous K order as frag_a_f32:
// lane L: col n = L%16 (key), half selects K range [half*16, half*16+16).
// B[kdim][n] = K[kr0+n][kdim] -> 16 contiguous floats of one K row.
__device__ __forceinline__ v16h frag_b_kT_f32(const float* __restrict__ kbase,
                                              int ld, int kr0, int c, int lane) {
  const int n = lane & 15, hf = lane >> 4;
  const float* p = kbase + (size_t)(kr0 + n) * ld + c * 32 + hf * 16;
  v16h r;
#pragma unroll
  for (int i = 0; i < 4; ++i) {
    v4f x = *(const v4f*)(p + 4 * i);
    r[4 * i + 0] = (_Float16)x[0];
    r[4 * i + 1] = (_Float16)x[1];
    r[4 * i + 2] = (_Float16)x[2];
    r[4 * i + 3] = (_Float16)x[3];
  }
  return r;
}

// ---- B-matrix fragment (32x16 f16) for B = V-tile: B[k][n] = V[kr0+k][d0+n].
// Lanes 0..15 read 16 consecutive floats of the same V row -> coalesced.
__device__ __forceinline__ v16h frag_b_v_f32(const float* __restrict__ vbase,
                                             int ld, int kr0, int d0, int lane) {
  const int n = lane & 15, hf = lane >> 4;
  const float* p = vbase + (size_t)(kr0 + hf * 16) * ld + d0 + n;
  v16h r;
#pragma unroll
  for (int j = 0; j < 16; ++j) r[j] = (_Float16)p[(size_t)j * ld];
  return r;
}

// ---- A-matrix fragment of P from per-wave LDS chunk (16x32 f16, row major).
// Two 16-byte ds loads per lane read K = half*16 + 0..15 contiguously.
__device__ __forceinline__ v16h frag_a_p(const _Float16* pb, int lane) {
  const int m = lane & 15, hf = lane >> 4;
  const _Float16* p = pb + m * 32 + hf * 16;
  v8h lo = *(const v8h*)(p);
  v8h hi = *(const v8h*)(p + 8);
  v16h r;
#pragma unroll
  for (int i = 0; i < 8; ++i) { r[i] = lo[i]; r[8 + i] = hi[i]; }
  return r;
}

// Merge two online-softmax states (m1,l1) <- (m1,l1) + (m2,l2).
__device__ __forceinline__ void sm_merge(float& m1, float& l1, float m2, float l2) {
  float mn = fmaxf(m1, m2);
  l1 = l1 * __expf(m1 - mn) + l2 * __expf(m2 - mn);
  m1 = mn;
}

__global__ __launch_bounds__(NTHREADS, 1)
void attn_fwd_kernel(const float* __restrict__ q, const float* __restrict__ k,
                     const float* __restrict__ v, float* __restrict__ out,
                     float* __restrict__ attn) {
  __shared__ float2 redml[16][NW];                    // cross-wave (m,l) per row
  __shared__ float obuf[16][Dn];                      // O accumulation (ds_add_f32)
  __shared__ __align__(16) _Float16 pbuf[NW][16][32]; // per-wave P restage

  const int tid  = threadIdx.x;
  const int wave = tid >> 5;
  const int lane = tid & 31;
  const int n    = lane & 15;   // column within 16-wide tile (C/D layout)
  const int hf   = lane >> 4;   // half selector: rows vv + 8*hf
  const int b    = blockIdx.y;
  const int q0   = blockIdx.x * 16;
  const int wbase = wave * WK;

  // zero the O accumulation buffer (atomically accumulated later)
  for (int i = tid; i < 16 * Dn; i += NTHREADS) ((float*)obuf)[i] = 0.0f;

  const float* qb = q + ((size_t)b * Nn + q0) * Dn;
  const float* kb = k + (size_t)b * Nn * Dn;
  const float* vb = v + (size_t)b * Nn * Dn;

  // ---- load Q A-fragments once (f32 -> f16), 4 chunks of K=32 over D=128 ----
  v16h qf[4];
#pragma unroll
  for (int c = 0; c < 4; ++c) qf[c] = frag_a_f32(qb, Dn, c, lane);

  // =======================================================================
  // Pass 1: S = Q*K^T tiles, online softmax state only (no S storage).
  // Per lane: m[vv], l[vv] for row (vv + 8*hf) over this lane's key subset.
  // =======================================================================
  float m8[8], l8[8];
#pragma unroll
  for (int vv = 0; vv < 8; ++vv) { m8[vv] = -3.0e38f; l8[vv] = 0.0f; }

#pragma unroll 2
  for (int t = 0; t < NT; ++t) {
    const int kr0 = wbase + t * 16;
    if (t + 1 < NT)  // speculative prefetch of next K tile (global_prefetch_b8)
      __builtin_prefetch(kb + (size_t)(kr0 + 16) * Dn + lane * 4, 0, 1);
    v8f acc = {};
#pragma unroll
    for (int c = 0; c < 4; ++c) {
      v16h bf = frag_b_kT_f32(kb, Dn, kr0, c, lane);
      acc = __builtin_amdgcn_wmma_f32_16x16x32_f16(
          false, qf[c], false, bf, (short)0, acc, false, false);
    }
#pragma unroll
    for (int vv = 0; vv < 8; ++vv) {
      float sv = acc[vv] * ATTN_SCALE;               // one value of row vv+8*hf
      float mn = fmaxf(m8[vv], sv);
      l8[vv] = l8[vv] * __expf(m8[vv] - mn) + __expf(sv - mn);
      m8[vv] = mn;
    }
  }

  // reduce (m,l) over the 16 lanes of each half (halves = disjoint rows)
#pragma unroll
  for (int vv = 0; vv < 8; ++vv)
#pragma unroll
    for (int off = 1; off <= 8; off <<= 1) {
      float om = __shfl_xor(m8[vv], off, 32);
      float ol = __shfl_xor(l8[vv], off, 32);
      sm_merge(m8[vv], l8[vv], om, ol);
    }
  if (n == 0) {
#pragma unroll
    for (int vv = 0; vv < 8; ++vv)
      redml[vv + 8 * hf][wave] = make_float2(m8[vv], l8[vv]);
  }
  __syncthreads();

  // cross-wave merge -> global max gm[vv] and inverse sum inv[vv]
  float gm[8], inv[8];
#pragma unroll
  for (int vv = 0; vv < 8; ++vv) {
    float mm = -3.0e38f, ll = 0.0f;
    for (int w = 0; w < NW; ++w) {
      float2 ml = redml[vv + 8 * hf][w];
      sm_merge(mm, ll, ml.x, ml.y);
    }
    gm[vv] = mm;
    inv[vv] = 1.0f / ll;
  }
  __syncthreads();

  // =======================================================================
  // Pass 2 (fused): recompute S per 32-key chunk, normalize, stream attn,
  // restage f16 P through LDS, and run GEMM2 on the chunk.
  // =======================================================================
  float* ab = attn + ((size_t)b * Nn + q0) * Nn + wbase;
  v8f o[8] = {};

#pragma unroll 1
  for (int cc = 0; cc < NT / 2; ++cc) {
#pragma unroll
    for (int tt = 0; tt < 2; ++tt) {
      const int t = cc * 2 + tt;
      const int kr0 = wbase + t * 16;
      v8f acc = {};
#pragma unroll
      for (int c = 0; c < 4; ++c) {
        v16h bf = frag_b_kT_f32(kb, Dn, kr0, c, lane);
        acc = __builtin_amdgcn_wmma_f32_16x16x32_f16(
            false, qf[c], false, bf, (short)0, acc, false, false);
      }
#pragma unroll
      for (int vv = 0; vv < 8; ++vv) {
        float pn = __expf(acc[vv] * ATTN_SCALE - gm[vv]) * inv[vv];
        ab[(size_t)(vv + 8 * hf) * Nn + t * 16 + n] = pn;        // attn store
        pbuf[wave][vv + 8 * hf][tt * 16 + n] = (_Float16)pn;     // P restage
      }
    }
    __syncthreads();
    v16h ap = frag_a_p(&pbuf[wave][0][0], lane);
    const int kr0 = wbase + cc * 32;
#pragma unroll
    for (int dt = 0; dt < 8; ++dt) {
      v16h bv = frag_b_v_f32(vb, Dn, kr0, dt * 16, lane);
      o[dt] = __builtin_amdgcn_wmma_f32_16x16x32_f16(
          false, ap, false, bv, (short)0, o[dt], false, false);
    }
    __syncthreads();
  }

  // ---- cross-wave reduction of partial O via LDS float atomics ----
#pragma unroll
  for (int dt = 0; dt < 8; ++dt)
#pragma unroll
    for (int vv = 0; vv < 8; ++vv)
      atomicAdd(&obuf[vv + 8 * hf][dt * 16 + n], o[dt][vv]);  // ds_add_f32
  __syncthreads();

  float* ob = out + ((size_t)b * Nn + q0) * Dn;
  for (int i = tid; i < 16 * Dn; i += NTHREADS) ob[i] = ((const float*)obuf)[i];
}

extern "C" void kernel_launch(void* const* d_in, const int* in_sizes, int n_in,
                              void* d_out, int out_size, void* d_ws, size_t ws_size,
                              hipStream_t stream) {
  (void)in_sizes; (void)n_in; (void)d_ws; (void)ws_size; (void)out_size;
  const float* q = (const float*)d_in[0];
  const float* k = (const float*)d_in[1];
  const float* v = (const float*)d_in[2];
  float* out  = (float*)d_out;                        // [B, N, D]
  float* attn = out + (size_t)Bn * Nn * Dn;           // [B, N, N]
  dim3 grid(Nn / 16, Bn);
  dim3 block(NTHREADS);
  hipLaunchKernelGGL(attn_fwd_kernel, grid, block, 0, stream, q, k, v, out, attn);
}
